// binary_conv_16415365005588
// MI455X (gfx1250) — matrix operand order repaired and checked by reading, not compiled, not run on hardware
//
#include <hip/hip_runtime.h>
#include <stdint.h>

typedef _Float16 v8h  __attribute__((ext_vector_type(8)));
typedef _Float16 v16h __attribute__((ext_vector_type(16)));
typedef float    v4f  __attribute__((ext_vector_type(4)));
typedef float    v8f  __attribute__((ext_vector_type(8)));

#define BB 64
#define HH 56
#define WW 56
#define CC 256
#define FF 256
#define KTAPS 9
#define MTILE 128
#define NTILE 128
#define KCHUNK 32
#define NCHUNKS (KTAPS * (CC / KCHUNK))   // 72

// ---------------------------------------------------------------------------
// Prep: fp16 round-trip + sign(clip(w)) -> f16 (+1 / -1 / 0), stored as
// row-major [K=2304][N=256] (HWIO is already (tap, c, f) = (k, n) order).
// ---------------------------------------------------------------------------
__global__ void bconv_prep_weights(const float* __restrict__ w,
                                   _Float16* __restrict__ w16, int n) {
  int i = blockIdx.x * 256 + threadIdx.x;
  if (i < n) {
    float q = (float)(_Float16)w[i];  // fp16 round-trip
    float s = (q > 0.0f) ? 1.0f : ((q < 0.0f) ? -1.0f : 0.0f);
    w16[i] = (_Float16)s;
  }
}

// ---------------------------------------------------------------------------
// Main: implicit-GEMM binary conv via V_WMMA_F32_16X16X32_F16.
// Block = 256 threads (8 waves). Tile: 128 (pixels) x 128 (filters).
// Double-buffered software pipeline: chunk c+1 is staged (A via
// global_load_async_to_lds_b128 / ASYNCcnt, B via coalesced read +
// transposed ds stores) while chunk c runs its 8 WMMAs. One barrier/chunk.
// ---------------------------------------------------------------------------
__launch_bounds__(256, 1)
__global__ void bconv_wmma_kernel(const float* __restrict__ x,
                                  const _Float16* __restrict__ w16,
                                  float* __restrict__ out) {
  __shared__ __align__(16) float    As32[2][MTILE * KCHUNK];  // 2 x 16 KB, m-major f32
  __shared__ __align__(32) _Float16 Bs[2][NTILE * KCHUNK];    // 2 x  8 KB, n-major f16

  const int tid   = threadIdx.x;
  const int wv    = tid >> 5;       // wave 0..7
  const int lane  = tid & 31;
  const int mBase = blockIdx.x * MTILE;   // pixel tile
  const int n0    = blockIdx.y * NTILE;   // filter tile

  // ---- A-staging coords: thread handles row rA, 16 f32 channels from cA ----
  const int rA = tid >> 1;            // 0..127
  const int cA = (tid & 1) * 16;      // 0 or 16 (f32 elements)
  const int p  = mBase + rA;          // linear output pixel
  const int b  = p / (HH * WW);
  const int hw = p - b * (HH * WW);
  const int h  = hw / WW;
  const int w  = hw - h * WW;
  const float* rowBase = x + ((size_t)b * HH * WW) * CC;  // image base for this row

  // ---- B-staging coords: thread handles k-row kB, 16 filters from nB ----
  const int kB = tid >> 3;            // 0..31
  const int nB = (tid & 7) * 16;      // 0..112

  // ---- Fragment coords (contiguous k order, same for A and B) ----
  const int mF  = wv * 16 + (lane & 15);  // A row within tile
  const int kbA = (lane >> 4) * 16;       // A: lanes 0-15 K=0..15, 16-31 K=16..31
  const int kbB = (lane >> 4) * 16;       // B: lanes 0-15 K=0..15, 16-31 K=16..31
  const int nF  = lane & 15;              // B/C column within 16x16 tile

  v8f acc[8];
#pragma unroll
  for (int j = 0; j < 8; ++j)
#pragma unroll
    for (int e = 0; e < 8; ++e) acc[j][e] = 0.0f;

  // Stage chunk c (flat index: tap = c>>3, kc = c&7) into buffer (c&1).
  auto stage = [&](int c) {
    const int bufS = c & 1;
    const int tapS = c >> 3;
    const int c0   = (c & 7) * KCHUNK;
    const int dh   = tapS / 3 - 1;
    const int dw   = tapS % 3 - 1;
    const int hh   = h + dh;
    const int ww   = w + dw;
    const bool val = (hh >= 0) && (hh < HH) && (ww >= 0) && (ww < WW);

    // ---- A: async DMA of raw f32 chunk into LDS (64 B / thread) ----
    if (val) {
      const float* srcA =
          rowBase + ((size_t)hh * WW + ww) * CC + c0 + cA;
      const uint64_t g = (uint64_t)(uintptr_t)srcA;
      const uint32_t l =
          (uint32_t)(uintptr_t)&As32[bufS][rA * KCHUNK + cA];
#pragma unroll
      for (int i = 0; i < 4; ++i) {
        asm volatile("global_load_async_to_lds_b128 %0, %1, off"
                     :: "v"(l + 16u * i), "v"(g + 16u * i)
                     : "memory");
      }
    } else {
      v4f z;
      z[0] = z[1] = z[2] = z[3] = 0.0f;
#pragma unroll
      for (int i = 0; i < 4; ++i)
        *(v4f*)&As32[bufS][rA * KCHUNK + cA + 4 * i] = z;
    }

    // ---- B: coalesced f16 row read, transposed (n-major) LDS write ----
    const _Float16* wrow =
        w16 + ((size_t)tapS * CC + c0 + kB) * FF + n0 + nB;
    v8h blo = *(const v8h*)(wrow);
    v8h bhi = *(const v8h*)(wrow + 8);
    __builtin_prefetch(wrow + (size_t)KCHUNK * FF, 0, 1);
#pragma unroll
    for (int e = 0; e < 8; ++e) {
      Bs[bufS][(nB + e) * KCHUNK + kB]     = blo[e];
      Bs[bufS][(nB + e + 8) * KCHUNK + kB] = bhi[e];
    }
  };

  // ---- pipeline prologue ----
  stage(0);
  asm volatile("s_wait_asynccnt 0x0" ::: "memory");
  __syncthreads();

  for (int c = 0; c < NCHUNKS; ++c) {
    const int buf = c & 1;

    // stage next chunk into the other buffer (overlaps with WMMA below)
    if (c + 1 < NCHUNKS) stage(c + 1);

    // ---- A fragment: read f32 from LDS in contiguous k order, cvt f16 ----
    v4f a0 = *(const v4f*)&As32[buf][mF * KCHUNK + kbA + 0];
    v4f a1 = *(const v4f*)&As32[buf][mF * KCHUNK + kbA + 4];
    v4f a2 = *(const v4f*)&As32[buf][mF * KCHUNK + kbA + 8];
    v4f a3 = *(const v4f*)&As32[buf][mF * KCHUNK + kbA + 12];
    v16h afrag;
#pragma unroll
    for (int e = 0; e < 4; ++e) {
      afrag[e + 0]  = (_Float16)a0[e];
      afrag[e + 4]  = (_Float16)a1[e];
      afrag[e + 8]  = (_Float16)a2[e];
      afrag[e + 12] = (_Float16)a3[e];
    }

    // ---- 8 WMMA per wave: wave's 16 M-rows vs all 8 N-subtiles ----
#pragma unroll
    for (int j = 0; j < 8; ++j) {
      v16h bfrag = *(const v16h*)&Bs[buf][(j * 16 + nF) * KCHUNK + kbB];
      acc[j] = __builtin_amdgcn_wmma_f32_16x16x32_f16(
          false, afrag, false, bfrag, (short)0, acc[j], false, false);
    }

    // drain this wave's async copies for chunk c+1, then handshake
    asm volatile("s_wait_asynccnt 0x0" ::: "memory");
    __syncthreads();
  }

  // ---- write C: VGPR i holds M=i (lanes 0-15) / M=8+i (lanes 16-31) ----
  const int mAdd   = (lane >> 4) * 8;
  const int pixRow = mBase + wv * 16 + mAdd;
#pragma unroll
  for (int j = 0; j < 8; ++j) {
    const int n = n0 + j * 16 + nF;
#pragma unroll
    for (int i = 0; i < 8; ++i) {
      out[(size_t)(pixRow + i) * FF + n] = acc[j][i];
    }
  }
}

// ---------------------------------------------------------------------------
extern "C" void kernel_launch(void* const* d_in, const int* in_sizes, int n_in,
                              void* d_out, int out_size, void* d_ws, size_t ws_size,
                              hipStream_t stream) {
  const float* x = (const float*)d_in[0];   // (64,56,56,256) f32
  const float* w = (const float*)d_in[1];   // (3,3,256,256)  f32
  float* out = (float*)d_out;               // (64,56,56,256) f32
  _Float16* w16 = (_Float16*)d_ws;          // 2304x256 f16 binarized

  const int nW = KTAPS * CC * FF;  // 589824
  bconv_prep_weights<<<(nW + 255) / 256, 256, 0, stream>>>(w, w16, nW);

  dim3 grid((BB * HH * WW) / MTILE, FF / NTILE, 1);  // (1568, 2)
  bconv_wmma_kernel<<<grid, 256, 0, stream>>>(x, w16, out);
}
